// Attention_2808908612304
// MI455X (gfx1250) — compile-verified, matrix operand order checked by reading, not run on hardware
//
#include <hip/hip_runtime.h>
#include <hip/hip_bf16.h>

// Problem sizes (fixed by the reference)
#define BATCH 512
#define NN    256   // N (num_vecs)
#define FF    512   // F (feature dim, K of GEMM1)
#define HH    512   // H (hidden dim)

#define FT_STRIDE 520            // bf16 elements per ft row in LDS (16B-aligned rows)
#define SMEM_BYTES 274448        // ftS(266240) + 6*1KB f32 arrays + bftS(2KB) + scalars

typedef __attribute__((ext_vector_type(16))) __bf16 v16bf;
typedef __attribute__((ext_vector_type(8)))  float  v8f;

struct __align__(16) BF16x8 { __bf16 v[8]; };
struct __align__(8)  BF16x4 { __bf16 v[4]; };

// ---------------------------------------------------------------------------
// Helpers
// ---------------------------------------------------------------------------

// Convert 8 consecutive f32 to bf16 into d[0..7]
__device__ __forceinline__ void cvt8(const float4 a, const float4 b, __bf16* d) {
    d[0] = (__bf16)a.x; d[1] = (__bf16)a.y; d[2] = (__bf16)a.z; d[3] = (__bf16)a.w;
    d[4] = (__bf16)b.x; d[5] = (__bf16)b.y; d[6] = (__bf16)b.z; d[7] = (__bf16)b.w;
}

// A/B fragment (16x32 bf16) from a row-major f32 matrix row, K contiguous.
// Element j -> k = k0 + j + (j>=8)*8 + hi*8  (per ISA 16-bit A layout; two 32B groups)
__device__ __forceinline__ v16bf load_frag_f32(const float* __restrict__ row, int k0, int hi) {
    const float4* p0 = (const float4*)(row + k0 + hi * 8);
    const float4* p1 = (const float4*)(row + k0 + 16 + hi * 8);
    union { v16bf v; __bf16 e[16]; } u;
    cvt8(p0[0], p0[1], &u.e[0]);
    cvt8(p1[0], p1[1], &u.e[8]);
    return u.v;
}

// Same fragment pattern from a pre-converted bf16 row (global, 2x b128 loads)
__device__ __forceinline__ v16bf load_frag_gbf(const __bf16* __restrict__ row, int k0, int hi) {
    union { v16bf v; BF16x8 h[2]; } u;
    u.h[0] = *(const BF16x8*)(row + k0 + hi * 8);
    u.h[1] = *(const BF16x8*)(row + k0 + 16 + hi * 8);
    return u.v;
}

// Same fragment pattern from a bf16 row in LDS (two ds_load_b128 groups)
__device__ __forceinline__ v16bf load_frag_lds(const __bf16* row, int k0, int hi) {
    union { v16bf v; BF16x8 h[2]; } u;
    u.h[0] = *(const BF16x8*)(row + k0 + hi * 8);
    u.h[1] = *(const BF16x8*)(row + k0 + 16 + hi * 8);
    return u.v;
}

// Fast tanh via the transcendental unit: (e^{2x}-1)/(e^{2x}+1), clamped so
// e^{2x} stays finite (avoids inf/inf NaN). ~6 VALU vs ~20+ for ocml tanhf.
__device__ __forceinline__ float fast_tanh(float x) {
    x = fminf(fmaxf(x, -15.f), 15.f);
    float e = __expf(2.f * x);
    return (e - 1.f) / (e + 1.f);
}

// ---------------------------------------------------------------------------
// Prep kernel: one-time f32 -> bf16 conversion of Wft (H x F) and Wf (N x H)
// into workspace. Weights are L2-resident and reused by all 512 workgroups;
// this halves B-side bytes and removes in-loop conversions.
// ---------------------------------------------------------------------------
extern "C" __global__ __launch_bounds__(256)
void prep_weights(const float* __restrict__ Wft, const float* __restrict__ Wf,
                  __bf16* __restrict__ wftB, __bf16* __restrict__ wfB)
{
    int i = blockIdx.x * blockDim.x + threadIdx.x;   // one float4 per thread
    const int nWft4 = (HH * FF) / 4;                 // 65536
    const int nWf4  = (NN * HH) / 4;                 // 32768
    if (i < nWft4) {
        float4 v = ((const float4*)Wft)[i];
        BF16x4 o; o.v[0] = (__bf16)v.x; o.v[1] = (__bf16)v.y;
                  o.v[2] = (__bf16)v.z; o.v[3] = (__bf16)v.w;
        ((BF16x4*)wftB)[i] = o;
    } else if (i < nWft4 + nWf4) {
        int j = i - nWft4;
        float4 v = ((const float4*)Wf)[j];
        BF16x4 o; o.v[0] = (__bf16)v.x; o.v[1] = (__bf16)v.y;
                  o.v[2] = (__bf16)v.z; o.v[3] = (__bf16)v.w;
        ((BF16x4*)wfB)[j] = o;
    }
}

// ---------------------------------------------------------------------------
// Fused attention kernel: one workgroup per batch, ft kept in 260KB of LDS.
// ---------------------------------------------------------------------------
extern "C" __global__ __launch_bounds__(512)
void attn_fused(const float* __restrict__ features, const float* __restrict__ hidden,
                const float* __restrict__ s_in,
                const __bf16* __restrict__ wftB, const float* __restrict__ bft,
                const __bf16* __restrict__ wfB,  const float* __restrict__ bfv,
                const float* __restrict__ Wh,  const float* __restrict__ bh,
                const float* __restrict__ Wc,  const float* __restrict__ bc,
                const float* __restrict__ Ws,  const float* __restrict__ bs,
                float* __restrict__ out)
{
    extern __shared__ char smem[];
    __bf16* ftS  = (__bf16*)smem;                          // [256][FT_STRIDE] bf16
    float*  hS   = (float*)(smem + 256 * FT_STRIDE * 2);   // [256] h = hidden@Wh^T + bh
    float*  bfS  = hS  + 256;                              // [256] bf bias
    float*  wcS  = bfS + 256;                              // [256] Wc row
    float*  zS   = wcS + 256;                              // [256] z accumulators
    float*  aS   = zS  + 256;                              // [256] softmax(z) (temp: ws dot)
    float*  redS = aS  + 256;                              // [256] reduction scratch
    float*  bftS = redS + 256;                             // [512] bft bias
    float*  scal = bftS + 512;                             // [0]=i_score [3]=b_gate

    const int b    = blockIdx.x;
    const int tid  = threadIdx.x;
    const int lane = tid & 31;
    const int wave = tid >> 5;
    const int hi   = (lane >> 4) & 1;
    const int ln   = lane & 15;
    const int wm   = wave >> 2;      // 4x4 wave grid, 64x64 tile per wave
    const int wn   = wave & 3;

    const float* featB = features + (size_t)b * NN * FF;

    // ---------------- Phase 0: small matvecs h, ws, i_score; stage biases ----
    bftS[tid] = bft[tid];
    if (tid < NN) {
        float acc = bh[tid];
        const float* wr = Wh + (size_t)tid * HH;
        const float* hv = hidden + (size_t)b * HH;
#pragma unroll 4
        for (int k = 0; k < HH; k += 4) {
            float4 w4 = *(const float4*)(wr + k);
            float4 x4 = *(const float4*)(hv + k);
            acc += w4.x * x4.x + w4.y * x4.y + w4.z * x4.z + w4.w * x4.w;
        }
        hS[tid]  = acc;
        bfS[tid] = bfv[tid];
        wcS[tid] = Wc[tid];
        zS[tid]  = bc[0];
    } else {
        int j = tid - NN;
        float acc = bs[j];
        const float* wr = Ws + (size_t)j * HH;
        const float* sv = s_in + (size_t)b * HH;
#pragma unroll 4
        for (int k = 0; k < HH; k += 4) {
            float4 w4 = *(const float4*)(wr + k);
            float4 x4 = *(const float4*)(sv + k);
            acc += w4.x * x4.x + w4.y * x4.y + w4.z * x4.z + w4.w * x4.w;
        }
        aS[j] = acc;                                      // ws (pre-tanh), temp in aS
    }
    __syncthreads();
    if (tid < NN) redS[tid] = fast_tanh(aS[tid] + hS[tid]) * wcS[tid];
    __syncthreads();
    for (int d = 128; d > 0; d >>= 1) {
        if (tid < d) redS[tid] += redS[tid + d];
        __syncthreads();
    }
    if (tid == 0) scal[0] = redS[0] + bc[0];              // i_score

    // ---------------- Phase 1: GEMM1  ft = relu(features @ Wft^T + bft) -------
    // M=256 (n), N=512 (h, two passes of 256), K=512 (f). bf16 WMMA, f32 acc.
    for (int p = 0; p < 2; ++p) {
        const int cb = p * 256 + wn * 64;
        v8f acc[16] = {};
        for (int k0 = 0; k0 < FF; k0 += 32) {
            v16bf afr[4], bfr[4];
#pragma unroll
            for (int t = 0; t < 4; ++t)
                afr[t] = load_frag_f32(featB + (size_t)(wm * 64 + t * 16 + ln) * FF, k0, hi);
#pragma unroll
            for (int t = 0; t < 4; ++t)
                bfr[t] = load_frag_gbf(wftB + (size_t)(cb + t * 16 + ln) * FF, k0, hi);
#pragma unroll
            for (int tm = 0; tm < 4; ++tm)
#pragma unroll
                for (int tn = 0; tn < 4; ++tn)
                    acc[tm * 4 + tn] = __builtin_amdgcn_wmma_f32_16x16x32_bf16(
                        false, afr[tm], false, bfr[tn], (short)0, acc[tm * 4 + tn],
                        false, false);
        }
        // epilogue: +bias, relu, bf16 store to LDS
#pragma unroll
        for (int tm = 0; tm < 4; ++tm)
#pragma unroll
            for (int tn = 0; tn < 4; ++tn) {
                const int col  = cb + tn * 16 + ln;
                const float bias = bftS[col];
#pragma unroll
                for (int r = 0; r < 8; ++r) {
                    const int row = wm * 64 + tm * 16 + r + hi * 8;
                    float v = acc[tm * 4 + tn][r] + bias;
                    ftS[row * FT_STRIDE + col] = (__bf16)(v > 0.f ? v : 0.f);
                }
            }
    }
    __syncthreads();

    // ---------------- Phase 2: GEMM2  f = ft @ Wf^T, fused tanh·Wc -> z -------
    // M=256 (n), N=256 (m2), K=512 (h). A from LDS, B from pre-converted Wf.
    {
        const int mb = wm * 64, nb = wn * 64;
        v8f acc[16] = {};
        for (int k0 = 0; k0 < HH; k0 += 32) {
            v16bf afr[4], bfr[4];
#pragma unroll
            for (int t = 0; t < 4; ++t)
                afr[t] = load_frag_lds(ftS + (mb + t * 16 + ln) * FT_STRIDE, k0, hi);
#pragma unroll
            for (int t = 0; t < 4; ++t)
                bfr[t] = load_frag_gbf(wfB + (size_t)(nb + t * 16 + ln) * HH, k0, hi);
#pragma unroll
            for (int tm = 0; tm < 4; ++tm)
#pragma unroll
                for (int tn = 0; tn < 4; ++tn)
                    acc[tm * 4 + tn] = __builtin_amdgcn_wmma_f32_16x16x32_bf16(
                        false, afr[tm], false, bfr[tn], (short)0, acc[tm * 4 + tn],
                        false, false);
        }
        // z[n] += sum_m tanh(f[n,m] + bf[m] + h[m]) * Wc[m]
#pragma unroll
        for (int tm = 0; tm < 4; ++tm)
#pragma unroll
            for (int r = 0; r < 8; ++r) {
                const int n_idx = mb + tm * 16 + r + hi * 8;
                float v = 0.f;
#pragma unroll
                for (int tn = 0; tn < 4; ++tn) {
                    const int m_idx = nb + tn * 16 + ln;
                    v += fast_tanh(acc[tm * 4 + tn][r] + bfS[m_idx] + hS[m_idx]) * wcS[m_idx];
                }
                v += __shfl_xor(v, 1, 16);
                v += __shfl_xor(v, 2, 16);
                v += __shfl_xor(v, 4, 16);
                v += __shfl_xor(v, 8, 16);
                if (ln == 0) atomicAdd(&zS[n_idx], v);    // ds_add_f32
            }
    }
    __syncthreads();

    // ---------------- Phase 3: softmax(z), b_gate from concat softmax ---------
    float myz = 0.f;
    if (tid < NN) { myz = zS[tid]; redS[tid] = myz; }
    __syncthreads();
    for (int d = 128; d > 0; d >>= 1) {
        if (tid < d) redS[tid] = fmaxf(redS[tid], redS[tid + d]);
        __syncthreads();
    }
    const float m1 = redS[0];
    __syncthreads();
    float e = 0.f;
    if (tid < NN) { e = __expf(myz - m1); redS[tid] = e; }
    __syncthreads();
    for (int d = 128; d > 0; d >>= 1) {
        if (tid < d) redS[tid] += redS[tid + d];
        __syncthreads();
    }
    const float S1 = redS[0];
    if (tid < NN) aS[tid] = e / S1;                       // a = softmax(z)
    if (tid == 0) {
        const float isc = scal[0];
        const float m2  = fmaxf(m1, isc);
        const float ei  = __expf(isc - m2);
        scal[3] = ei / (S1 * __expf(m1 - m2) + ei);       // b_gate (concat softmax tail)
    }
    __syncthreads();

    // ---------------- Phase 4: c = a @ ft, out = g*s + (1-g)*c ---------------
    const float g = scal[3];
    float c = 0.f;
    const __bf16* colp = ftS + tid;
#pragma unroll 4
    for (int n = 0; n < NN; ++n)
        c += aS[n] * (float)colp[n * FT_STRIDE];
    out[(size_t)b * HH + tid] = g * s_in[(size_t)b * HH + tid] + (1.f - g) * c;
}

extern "C" void kernel_launch(void* const* d_in, const int* in_sizes, int n_in,
                              void* d_out, int out_size, void* d_ws, size_t ws_size,
                              hipStream_t stream) {
    const float* features = (const float*)d_in[0];
    const float* hidden   = (const float*)d_in[1];
    // d_in[2] = inputs, d_in[3] = memory  -> unused by the reference
    const float* s_in     = (const float*)d_in[4];
    const float* Wft      = (const float*)d_in[5];
    const float* bft      = (const float*)d_in[6];
    const float* Wf       = (const float*)d_in[7];
    const float* bfv      = (const float*)d_in[8];
    const float* Wh       = (const float*)d_in[9];
    const float* bh       = (const float*)d_in[10];
    const float* Wc       = (const float*)d_in[11];
    const float* bc       = (const float*)d_in[12];
    const float* Ws       = (const float*)d_in[13];
    const float* bs       = (const float*)d_in[14];
    float* outp = (float*)d_out;

    // Workspace: bf16 copies of Wft (512x512) and Wf (256x512)
    __bf16* wftB = (__bf16*)d_ws;                 // 524288 bytes
    __bf16* wfB  = wftB + (size_t)HH * FF;        // 262144 bytes

    // Allow >64KB dynamic LDS (CDNA5 WGP: up to 320KB per workgroup).
    // Called unconditionally every launch (deterministic, no static state).
    hipFuncSetAttribute((const void*)attn_fused,
                        hipFuncAttributeMaxDynamicSharedMemorySize, SMEM_BYTES);

    const int prep_threads = (HH * FF + NN * HH) / 4;     // 98304
    prep_weights<<<(prep_threads + 255) / 256, 256, 0, stream>>>(Wft, Wf, wftB, wfB);

    attn_fused<<<BATCH, 512, SMEM_BYTES, stream>>>(
        features, hidden, s_in, wftB, bft, wfB, bfv, Wh, bh, Wc, bc, Ws, bs, outp);
}
